// EdgeBlock_22797686407604
// MI455X (gfx1250) — matrix operand order repaired and checked by reading, not compiled, not run on hardware
//
#include <hip/hip_runtime.h>

#define B_SZ   8
#define FIN    64
#define FOUT   128
#define FMID   64
#define NPTS   4096
#define KNN    20
#define TILE_P 8
#define EDGES  (TILE_P*KNN)     /* 160 */
#define NEG_SLOPE 0.01f

typedef __attribute__((ext_vector_type(16))) __bf16 v16bf;
typedef __attribute__((ext_vector_type(8)))  __bf16 v8bf;
typedef __attribute__((ext_vector_type(8)))  float  v8f;
typedef __attribute__((ext_vector_type(2)))  float  v2f;
typedef __attribute__((ext_vector_type(4)))  unsigned tdm_g0_t;
typedef __attribute__((ext_vector_type(8)))  int      tdm_g1_t;
typedef __attribute__((ext_vector_type(4)))  int      tdm_g2_t;
typedef __attribute__((ext_vector_type(8)))  int      tdm_g4_t;

union U16bf { v16bf v; v8bf h[2]; };

#if defined(__has_builtin)
#  if __has_builtin(__builtin_amdgcn_wmma_f32_16x16x4_f32)
#    define HAVE_WMMA_F32X4 1
#  endif
#  if __has_builtin(__builtin_amdgcn_tensor_load_to_lds) && \
      __has_builtin(__builtin_amdgcn_s_wait_tensorcnt)
#    define HAVE_TDM 1
#  endif
#endif
#ifndef HAVE_WMMA_F32X4
#  define HAVE_WMMA_F32X4 0
#endif
#ifndef HAVE_TDM
#  define HAVE_TDM 0
#endif

// ---------------- workspace layout (bytes) ----------------
#define OFF_SQN 0u
#define OFF_IDX (OFF_SQN + (unsigned)(B_SZ*NPTS*4))            // sq norms
#define OFF_W1B (OFF_IDX + (unsigned)(B_SZ*NPTS*KNN*4))        // knn indices
#define OFF_W2B (OFF_W1B + (unsigned)(FMID*FIN*2))
#define OFF_WXB (OFF_W2B + (unsigned)(FOUT*FMID*2))
#define OFF_WOB (OFF_WXB + (unsigned)(FOUT*2*FIN*2))
#define OFF_FP  (OFF_WOB + (unsigned)(FOUT*FOUT*KNN*2))
// fp floats: s1[64] c1[64] s2[128] c2[128] sx[128] cx[128]

// ---- CDNA5 async global->LDS (ASYNCcnt) -------------------------------
// the VDST VGPR holds the LDS byte address; generic shared pointers carry the LDS
// byte address in their low 32 bits (ISA 10.2 aperture rules).
__device__ __forceinline__ void async_g2l_b128(void* lds, const void* g) {
  unsigned ldsoff = (unsigned)(unsigned long long)lds;
  asm volatile("global_load_async_to_lds_b128 %0, %1, off"
               :: "v"(ldsoff), "v"(g) : "memory");
}
__device__ __forceinline__ void wait_async0() {
  asm volatile("s_wait_asynccnt 0x0" ::: "memory");
}

// ---- WMMA bf16 fragment loaders (contiguous k order: k = 16*(lane/16) + s) ----
__device__ __forceinline__ v16bf ld_afrag(const __bf16* base, int stride,
                                          int row0, int k0, int lane) {
  int r  = row0 + (lane & 15);
  int kb = k0 + ((lane & 16) ? 16 : 0);
  const __bf16* p = base + r * stride + kb;
  U16bf u;
  u.h[0] = *(const v8bf*)(p);
  u.h[1] = *(const v8bf*)(p + 8);
  return u.v;
}
__device__ __forceinline__ v16bf ld_bfrag(const __bf16* base, int stride,
                                          int col0, int k0, int lane) {
  int c  = col0 + (lane & 15);
  int kb = k0 + ((lane & 16) ? 16 : 0);
  return *(const v16bf*)(base + c * stride + kb);
}

// ---------------- kernel 0: weight prep ----------------
__global__ __launch_bounds__(256) void prep_kernel(
    const float* w1, const float* b1, const float* g1, const float* be1,
    const float* m1, const float* v1,
    const float* w2, const float* b2, const float* g2, const float* be2,
    const float* m2, const float* v2,
    const float* wx, const float* bx, const float* gx, const float* bex,
    const float* mx, const float* vx,
    const float* wout,
    __bf16* w1b, __bf16* w2b, __bf16* wxb, __bf16* wob, float* fp) {
  int tid = blockIdx.x * blockDim.x + threadIdx.x;
  int stp = gridDim.x * blockDim.x;
  for (int i = tid; i < FMID*FIN;  i += stp) w1b[i] = (__bf16)w1[i];
  for (int i = tid; i < FOUT*FMID; i += stp) w2b[i] = (__bf16)w2[i];
  for (int i = tid; i < FOUT*2*FIN; i += stp) wxb[i] = (__bf16)wx[i];
  // woutT[o][k*128 + i] = wout[o][i][k]
  for (int i = tid; i < FOUT*FOUT*KNN; i += stp) {
    int o = i / (FOUT*KNN), r = i % (FOUT*KNN);
    int kk = r >> 7, ii = r & 127;
    wob[i] = (__bf16)wout[(o*FOUT + ii)*KNN + kk];
  }
  for (int i = tid; i < FMID; i += stp) {
    float s = g1[i] * rsqrtf(v1[i] + 1e-5f);
    fp[i] = s;  fp[64 + i] = (b1[i] - m1[i]) * s + be1[i];
  }
  for (int i = tid; i < FOUT; i += stp) {
    float s2 = g2[i] * rsqrtf(v2[i] + 1e-5f);
    fp[128 + i] = s2;  fp[256 + i] = (b2[i] - m2[i]) * s2 + be2[i];
    float sx = gx[i] * rsqrtf(vx[i] + 1e-5f);
    fp[384 + i] = sx;  fp[512 + i] = (bx[i] - mx[i]) * sx + bex[i];
  }
}

// ---------------- kernel 1: squared norms ----------------
__global__ __launch_bounds__(256) void sqnorm_kernel(const float* __restrict__ x,
                                                     float* __restrict__ sqn) {
  int gid = blockIdx.x * 256 + threadIdx.x;       // over B*N
  int b = gid >> 12, n = gid & (NPTS - 1);
  float s = 0.f;
  #pragma unroll
  for (int f = 0; f < FIN; ++f) {
    float v = x[(b*FIN + f)*NPTS + n];
    s += v * v;
  }
  sqn[gid] = s;
}

// ------ kernel 2: kNN (TDM/async tile staging + fp32 WMMA Gram + top-20) ----
// LDS tiles are f-major so one 16B global beat == one 16B LDS beat.
__global__ __launch_bounds__(256) void knn_kernel(const float* __restrict__ x,
                                                  const float* __restrict__ sqn,
                                                  int* __restrict__ idxout) {
  extern __shared__ char smem[];
  float* qf  = (float*)(smem);             // [64][16]  (f-major)
  float* sqq = (float*)(smem + 4096);      // [16]
  float* sqc = (float*)(smem + 4160);      // [128]
  float* cf  = (float*)(smem + 4672);      // [64][128] (f-major)
  float* dch = (float*)(smem + 37440);     // [16][128]
  // merge region overlays cf/dch after the chunk loop:
  float* md  = (float*)(smem + 4672);      // [16][320]
  int*   mi  = (int*)  (smem + 25152);     // [16][320]

  int tid = threadIdx.x, lane = tid & 31, wid = tid >> 5;
  int b  = blockIdx.x >> 8;
  int nb = (blockIdx.x & 255) * 16;

  // query tile 16x64 via async global->LDS b128 (one 16B beat per thread)
  {
    int f = tid >> 2, q4 = (tid & 3) * 4;
    async_g2l_b128(&qf[f*16 + q4], &x[(b*FIN + f)*NPTS + nb + q4]);
  }
  if (tid < 16) sqq[tid] = sqn[b*NPTS + nb + tid];
  wait_async0();

  float bd[20]; int bi[20];
  #pragma unroll
  for (int j = 0; j < 20; ++j) { bd[j] = 3.0e38f; bi[j] = 0; }
  int qrow  = tid >> 4;
  int ssel  = tid & 15;
  int nglob = nb + qrow;

  for (int cb = 0; cb < NPTS; cb += 128) {
    __syncthreads();
#if HAVE_TDM
    if (wid == 0) {
      // D#: 2-D tensor x[b] (dim0=4096 contiguous, 64 rows), tile 128x64
      unsigned long long ga =
          (unsigned long long)(const void*)(x + (unsigned long long)b*FIN*NPTS + cb);
      unsigned ldsa = (unsigned)(unsigned long long)(void*)cf;
      tdm_g0_t g0;
      g0[0] = 1u;                                     // count=1, user desc
      g0[1] = ldsa;                                   // lds_addr (bytes)
      g0[2] = (unsigned)(ga & 0xffffffffu);           // global_addr[31:0]
      g0[3] = (unsigned)((ga >> 32) & 0x01ffffffu)    // global_addr[56:32]
              | (2u << 30);                           // type=2 ("image")
      tdm_g1_t g1;
      g1[0] = (int)(2u << 16);                        // data_size=4B, no mask
      g1[1] = (int)(((unsigned)NPTS & 0xffffu) << 16);// tensor_dim0[15:0]
      g1[2] = (int)(((unsigned)FIN) << 16);           // dim0 hi=0 | tensor_dim1 lo
      g1[3] = (int)(128u << 16);                      // dim1 hi=0 | tile_dim0=128
      g1[4] = (int)(64u);                             // tile_dim1=64, tile_dim2=0
      g1[5] = (int)NPTS;                              // tensor_dim0_stride lo32
      g1[6] = 0;                                      // stride0 hi | stride1 lo
      g1[7] = 0;                                      // stride1 hi
      tdm_g2_t gz = {0, 0, 0, 0};
      tdm_g4_t gz8 = {0, 0, 0, 0, 0, 0, 0, 0};
      __builtin_amdgcn_tensor_load_to_lds(g0, g1, gz, gz, gz8, 0);
      __builtin_amdgcn_s_wait_tensorcnt(0);
    }
#else
    // fallback: per-lane async b128 copies (2048 beats / chunk)
    #pragma unroll
    for (int it = 0; it < 8; ++it) {
      int i = tid + it*256;
      int f = i >> 5, mq = (i & 31) * 4;
      async_g2l_b128(&cf[f*128 + mq], &x[(b*FIN + f)*NPTS + cb + mq]);
    }
    wait_async0();
#endif
    if (tid < 128) sqc[tid] = sqn[b*NPTS + cb + tid];
    __syncthreads();

#if HAVE_WMMA_F32X4
    {
      v8f acc = {};
      #pragma unroll
      for (int s = 0; s < 16; ++s) {       // K = 64 in steps of 4
        int kb = s*4 + ((lane & 16) ? 2 : 0);
        int m  = lane & 15;
        int cl = wid*16 + m;
        v2f av, bv;
        av.x = qf[kb*16 + m];
        av.y = qf[(kb + 1)*16 + m];
        bv.x = cf[kb*128 + cl];
        bv.y = cf[(kb + 1)*128 + cl];
        acc = __builtin_amdgcn_wmma_f32_16x16x4_f32(
            false, av, false, bv, (short)0, acc, false, false);
      }
      int col = wid*16 + (lane & 15);
      #pragma unroll
      for (int jj = 0; jj < 8; ++jj) {
        int row = jj + ((lane & 16) ? 8 : 0);
        dch[row*128 + col] = sqq[row] + sqc[col] - 2.0f*acc[jj];
      }
    }
#else
    for (int t = tid; t < 16*128; t += 256) {
      int row = t >> 7, col = t & 127;
      float s = 0.f;
      for (int f = 0; f < FIN; ++f) s += qf[f*16 + row] * cf[f*128 + col];
      dch[t] = sqq[row] + sqc[col] - 2.f*s;
    }
#endif
    __syncthreads();

    for (int c = ssel; c < 128; c += 16) {
      float d = dch[qrow*128 + c];
      int ci = cb + c;
      if (ci != nglob && d < bd[19]) {
        bd[19] = d; bi[19] = ci;
        #pragma unroll
        for (int j = 19; j > 0; --j)
          if (bd[j] < bd[j-1]) {
            float tf = bd[j]; bd[j] = bd[j-1]; bd[j-1] = tf;
            int   ti = bi[j]; bi[j] = bi[j-1]; bi[j-1] = ti;
          }
      }
    }
  }
  __syncthreads();
  #pragma unroll
  for (int j = 0; j < 20; ++j) {
    md[qrow*320 + ssel*20 + j] = bd[j];
    mi[qrow*320 + ssel*20 + j] = bi[j];
  }
  __syncthreads();
  if (ssel == 0) {
    int res[20];
    #pragma unroll
    for (int slot = 0; slot < 20; ++slot) {
      float best = 3.0e38f; int bj = 0;
      for (int t = 0; t < 320; ++t) {
        float d = md[qrow*320 + t];
        if (d < best) { best = d; bj = t; }
      }
      res[slot] = mi[qrow*320 + bj];
      md[qrow*320 + bj] = 3.0e38f;
    }
    #pragma unroll
    for (int j = 0; j < 20; ++j)
      idxout[(b*NPTS + nglob)*KNN + j] = res[j];
  }
}

// ------- kernel 3: fused edge MLP + attention + contraction (bf16 WMMA) -----
__global__ __launch_bounds__(256) void fused_edge_mlp(
    const float* __restrict__ x, const int* __restrict__ nidx,
    const __bf16* __restrict__ w1b, const __bf16* __restrict__ w2b,
    const __bf16* __restrict__ wxb, const __bf16* __restrict__ wob,
    const float* __restrict__ fp, const float* __restrict__ bout,
    float* __restrict__ out) {
  extern __shared__ char smem[];
  __bf16* E  = (__bf16*)(smem);            // [160][128] edge feats (col-major)
  __bf16* T1 = (__bf16*)(smem + 40960);    // [160][64]
  __bf16* W  = (__bf16*)(smem + 61440);    // [160][128] logits -> softmax wts
  __bf16* HW = (__bf16*)(smem + 102400);   // [160][128] h*w
  int*   nid = (int*)  (smem + 143360);    // [160]

  int tid = threadIdx.x, lane = tid & 31, wid = tid >> 5;
  int b  = blockIdx.x >> 9;
  int nb = (blockIdx.x & 511) * TILE_P;

  for (int t = tid; t < EDGES; t += 256)
    nid[t] = nidx[(b*NPTS + nb + t/KNN)*KNN + (t % KNN)];
  __syncthreads();

  // build edges e = n_local*20 + k: [central(64) | nbr-central(64)]
  for (int f = 0; f < FIN; ++f) {
    for (int e = tid; e < EDGES; e += 256) {
      int n = nb + e / KNN;
      float c  = x[(b*FIN + f)*NPTS + n];
      float nv = x[(b*FIN + f)*NPTS + nid[e]];
      E[e*128 + f]      = (__bf16)c;
      E[e*128 + 64 + f] = (__bf16)(nv - c);
    }
  }
  __syncthreads();

  // GEMM1: T1 = lrelu(bn(W1 @ diff)), M=64 K=64 N=160
  for (int j = wid; j < 40; j += 8) {
    int ot = j / 10, et = (j % 10) * 16;
    v8f acc = {};
    #pragma unroll
    for (int s = 0; s < 2; ++s) {
      v16bf a  = ld_afrag(w1b, FIN, ot*16, s*32, lane);
      v16bf bm = ld_bfrag(E, 128, et, 64 + s*32, lane);
      acc = __builtin_amdgcn_wmma_f32_16x16x32_bf16(false, a, false, bm,
                                                    (short)0, acc, false, false);
    }
    int col = et + (lane & 15);
    int rb  = ot*16 + ((lane & 16) ? 8 : 0);
    #pragma unroll
    for (int jj = 0; jj < 8; ++jj) {
      int ch = rb + jj;
      float t = acc[jj]*fp[ch] + fp[64 + ch];
      t = t >= 0.f ? t : NEG_SLOPE*t;
      T1[col*64 + ch] = (__bf16)t;
    }
  }
  __syncthreads();

  // GEMM2: W = lrelu(bn(W2 @ T1)), M=128 K=64 N=160
  for (int j = wid; j < 80; j += 8) {
    int ot = j / 10, et = (j % 10) * 16;
    v8f acc = {};
    #pragma unroll
    for (int s = 0; s < 2; ++s) {
      v16bf a  = ld_afrag(w2b, FMID, ot*16, s*32, lane);
      v16bf bm = ld_bfrag(T1, 64, et, s*32, lane);
      acc = __builtin_amdgcn_wmma_f32_16x16x32_bf16(false, a, false, bm,
                                                    (short)0, acc, false, false);
    }
    int col = et + (lane & 15);
    int rb  = ot*16 + ((lane & 16) ? 8 : 0);
    #pragma unroll
    for (int jj = 0; jj < 8; ++jj) {
      int ch = rb + jj;
      float t = acc[jj]*fp[128 + ch] + fp[256 + ch];
      t = t >= 0.f ? t : NEG_SLOPE*t;
      W[col*128 + ch] = (__bf16)t;
    }
  }
  __syncthreads();

  // softmax over k per (channel o, point n), in place in W
  for (int t = tid; t < FOUT*TILE_P; t += 256) {
    int o = t & 127, n = t >> 7;
    int base = (n*KNN)*128 + o;
    float mxv = -3.0e38f;
    #pragma unroll
    for (int kk = 0; kk < KNN; ++kk) mxv = fmaxf(mxv, (float)W[base + kk*128]);
    float ev[KNN]; float s = 0.f;
    #pragma unroll
    for (int kk = 0; kk < KNN; ++kk) {
      float e2 = expf((float)W[base + kk*128] - mxv);
      ev[kk] = e2; s += e2;
    }
    float inv = 1.f / s;
    #pragma unroll
    for (int kk = 0; kk < KNN; ++kk) W[base + kk*128] = (__bf16)(ev[kk]*inv);
  }
  __syncthreads();

  // GEMM3: h = lrelu(bn(Wx @ E)); HW = h * w, M=128 K=128 N=160
  for (int j = wid; j < 80; j += 8) {
    int ot = j / 10, et = (j % 10) * 16;
    v8f acc = {};
    #pragma unroll
    for (int s = 0; s < 4; ++s) {
      v16bf a  = ld_afrag(wxb, 2*FIN, ot*16, s*32, lane);
      v16bf bm = ld_bfrag(E, 128, et, s*32, lane);
      acc = __builtin_amdgcn_wmma_f32_16x16x32_bf16(false, a, false, bm,
                                                    (short)0, acc, false, false);
    }
    int col = et + (lane & 15);
    int rb  = ot*16 + ((lane & 16) ? 8 : 0);
    #pragma unroll
    for (int jj = 0; jj < 8; ++jj) {
      int ch = rb + jj;
      float h = acc[jj]*fp[384 + ch] + fp[512 + ch];
      h = h >= 0.f ? h : NEG_SLOPE*h;
      float wv = (float)W[col*128 + ch];
      HW[col*128 + ch] = (__bf16)(h*wv);
    }
  }
  __syncthreads();

  // GEMM4: out[o,n] = sum_c woutT[o][c] * HWflat[n][c], c=k*128+i, K=2560
  {
    int ot  = wid;                          // 8 waves <-> 8 o-tiles
    int nco = lane & 15; if (nco >= TILE_P) nco -= TILE_P;  // clamp dummy cols
    v8f acc = {};
    for (int s = 0; s < 80; ++s) {
      v16bf a = ld_afrag(wob, FOUT*KNN, ot*16, s*32, lane);
      const __bf16* p = HW + nco*(KNN*128) + s*32 + ((lane & 16) ? 16 : 0);
      v16bf bm = *(const v16bf*)p;
      acc = __builtin_amdgcn_wmma_f32_16x16x32_bf16(false, a, false, bm,
                                                    (short)0, acc, false, false);
    }
    int ncol = lane & 15;
    if (ncol < TILE_P) {
      int rb = ot*16 + ((lane & 16) ? 8 : 0);
      #pragma unroll
      for (int jj = 0; jj < 8; ++jj) {
        int o = rb + jj;
        out[(b*FOUT + o)*NPTS + nb + ncol] = acc[jj] + bout[o];
      }
    }
  }
}

extern "C" void kernel_launch(void* const* d_in, const int* in_sizes, int n_in,
                              void* d_out, int out_size, void* d_ws, size_t ws_size,
                              hipStream_t stream) {
  (void)in_sizes; (void)n_in; (void)out_size; (void)ws_size;
  const float* x    = (const float*)d_in[0];
  const float* w1   = (const float*)d_in[1];
  const float* b1   = (const float*)d_in[2];
  const float* g1   = (const float*)d_in[3];
  const float* be1  = (const float*)d_in[4];
  const float* m1   = (const float*)d_in[5];
  const float* v1   = (const float*)d_in[6];
  const float* w2   = (const float*)d_in[7];
  const float* b2   = (const float*)d_in[8];
  const float* g2   = (const float*)d_in[9];
  const float* be2  = (const float*)d_in[10];
  const float* m2   = (const float*)d_in[11];
  const float* v2   = (const float*)d_in[12];
  const float* wx   = (const float*)d_in[13];
  const float* bx   = (const float*)d_in[14];
  const float* gx   = (const float*)d_in[15];
  const float* bex  = (const float*)d_in[16];
  const float* mx   = (const float*)d_in[17];
  const float* vx   = (const float*)d_in[18];
  const float* wout = (const float*)d_in[19];
  const float* bout = (const float*)d_in[20];

  char* ws = (char*)d_ws;
  float*  sqn = (float*)(ws + OFF_SQN);
  int*    idx = (int*)  (ws + OFF_IDX);
  __bf16* w1b = (__bf16*)(ws + OFF_W1B);
  __bf16* w2b = (__bf16*)(ws + OFF_W2B);
  __bf16* wxb = (__bf16*)(ws + OFF_WXB);
  __bf16* wob = (__bf16*)(ws + OFF_WOB);
  float*  fp  = (float*)(ws + OFF_FP);

  prep_kernel<<<320, 256, 0, stream>>>(w1,b1,g1,be1,m1,v1, w2,b2,g2,be2,m2,v2,
                                       wx,bx,gx,bex,mx,vx, wout,
                                       w1b, w2b, wxb, wob, fp);
  sqnorm_kernel<<<(B_SZ*NPTS)/256, 256, 0, stream>>>(x, sqn);
  knn_kernel<<<B_SZ*(NPTS/16), 256, 45632, stream>>>(x, sqn, idx);
  fused_edge_mlp<<<B_SZ*(NPTS/TILE_P), 256, 144000, stream>>>(
      x, idx, w1b, w2b, wxb, wob, fp, bout, (float*)d_out);
}
